// DeformableTransformerEncoderLayer_22969485099114
// MI455X (gfx1250) — compile-verified, matrix operand order checked by reading, not run on hardware
//
#include <hip/hip_runtime.h>
#include <hip/hip_bf16.h>

typedef _Float16 f16;
typedef __attribute__((ext_vector_type(16))) _Float16 v16h;
typedef __attribute__((ext_vector_type(8)))  float    v8f;

#define NH      8
#define CMODEL  256
#define DHEAD   32
#define DFFN    1024
#define LQ      21760
#define NB      2
#define MROWS   (NB * LQ)      // 43520 rows (tokens)

union Frag { v16h v; unsigned int p[8]; };

__device__ __forceinline__ f16 f2h(float x) { return (f16)x; }

// ---------------------------------------------------------------------------
// Generic WMMA GEMM:  out[M,N] = A[M,K](f16) * W[N,K](f16)^T + bias[N]
// Wave tile 32x64 (2 A-frags x 4 B-frags -> 8 wmma per 32-wide K chunk).
// Block = 8 waves -> 128x128 tile. Requires M%128==0, N%128==0... N%128? we
// launch grid.y = N/128; all N used (128,256,1024) qualify. K%32==0.
// ---------------------------------------------------------------------------
__global__ __launch_bounds__(256) void gemm_f16_kernel(
    const f16* __restrict__ A, const f16* __restrict__ W,
    const float* __restrict__ bias,
    float* __restrict__ outF, f16* __restrict__ outH,
    int M, int N, int K, int relu)
{
  const int lane = threadIdx.x;
  const int wave = threadIdx.y;
  const int half = lane >> 4;
  const int mr   = lane & 15;
  const int tileM = blockIdx.x * 128 + (wave >> 1) * 32;
  const int tileN = blockIdx.y * 128 + (wave & 1) * 64;

  v8f acc[2][4] = {};

  const f16* A0 = A + (size_t)(tileM + mr) * K;
  const f16* A1 = A + (size_t)(tileM + 16 + mr) * K;

  for (int k0 = 0; k0 < K; k0 += 32) {
    Frag a0, a1;
#pragma unroll
    for (int j = 0; j < 8; ++j) {
      const int kk = k0 + ((j & 4) << 2) + (half << 3) + ((j & 3) << 1);
      a0.p[j] = *reinterpret_cast<const unsigned int*>(A0 + kk);
      a1.p[j] = *reinterpret_cast<const unsigned int*>(A1 + kk);
    }
#pragma unroll
    for (int t = 0; t < 4; ++t) {
      const f16* Wr = W + (size_t)(tileN + t * 16 + mr) * K;
      Frag bfr;
#pragma unroll
      for (int j = 0; j < 8; ++j) {
        const int kk = k0 + ((j & 4) << 2) + (half << 3) + ((j & 3) << 1);
        bfr.p[j] = *reinterpret_cast<const unsigned int*>(Wr + kk);
      }
      acc[0][t] = __builtin_amdgcn_wmma_f32_16x16x32_f16(
          false, a0.v, false, bfr.v, (short)0, acc[0][t], false, false);
      acc[1][t] = __builtin_amdgcn_wmma_f32_16x16x32_f16(
          false, a1.v, false, bfr.v, (short)0, acc[1][t], false, false);
    }
  }

#pragma unroll
  for (int i = 0; i < 2; ++i)
#pragma unroll
    for (int t = 0; t < 4; ++t)
#pragma unroll
      for (int r = 0; r < 8; ++r) {
        const int m = tileM + i * 16 + half * 8 + r;
        const int n = tileN + t * 16 + mr;
        float x = acc[i][t][r] + bias[n];
        if (relu) x = x > 0.f ? x : 0.f;
        const size_t o = (size_t)m * N + n;
        if (outF) outF[o] = x;
        if (outH) outH[o] = f2h(x);
      }
}

// ---------------------------------------------------------------------------
// Elementwise helpers
// ---------------------------------------------------------------------------
__global__ void f32_to_f16_kernel(const float* __restrict__ s, f16* __restrict__ d, int n)
{
  int i = blockIdx.x * blockDim.x + threadIdx.x;
  if (i < n) d[i] = (f16)s[i];
}

__global__ void prep_kernel(const float* __restrict__ src, const float* __restrict__ pos,
                            f16* __restrict__ qh, f16* __restrict__ sh, int n)
{
  int i = blockIdx.x * blockDim.x + threadIdx.x;
  if (i < n) { float s = src[i]; qh[i] = (f16)(s + pos[i]); sh[i] = (f16)s; }
}

// softmax over L*P=16 logits per (b,q,h); aw layout [MROWS,128]
__global__ void softmax16_kernel(float* __restrict__ aw)
{
  const int i = blockIdx.x * blockDim.x + threadIdx.x;
  if (i >= MROWS * NH) return;
  float* p = aw + (size_t)(i >> 3) * 128 + (size_t)(i & 7) * 16;
  float mx = p[0];
#pragma unroll
  for (int k = 1; k < 16; ++k) mx = fmaxf(mx, p[k]);
  float e[16]; float sum = 0.f;
#pragma unroll
  for (int k = 0; k < 16; ++k) { e[k] = expf(p[k] - mx); sum += e[k]; }
  const float inv = 1.f / sum;
#pragma unroll
  for (int k = 0; k < 16; ++k) p[k] = e[k] * inv;
}

// ---------------------------------------------------------------------------
// Deformable-attention bilinear sampling. One wave per (b,q,h); lane = d.
// v: [B, LQ, 256] f32 (channel = h*32+d); out attn: [MROWS, 256] f16.
// ---------------------------------------------------------------------------
__device__ __forceinline__ float gatherv(const float* __restrict__ vb, int s,
                                         int Wl, int Hl, int xi, int yi)
{
  const bool valid = (xi >= 0) & (xi < Wl) & (yi >= 0) & (yi < Hl);
  const int xc = xi < 0 ? 0 : (xi > Wl - 1 ? Wl - 1 : xi);
  const int yc = yi < 0 ? 0 : (yi > Hl - 1 ? Hl - 1 : yi);
  const float val = vb[(size_t)(s + yc * Wl + xc) * CMODEL];
  return valid ? val : 0.f;
}

__global__ __launch_bounds__(256) void sample_kernel(
    const float* __restrict__ v, const float* __restrict__ offv,
    const float* __restrict__ aw, const float* __restrict__ ref,
    f16* __restrict__ attn)
{
  const int item = blockIdx.x * 8 + threadIdx.y;   // (b*Q)*H + h
  const int lane = threadIdx.x;                    // d in [0,32)
  const int bq = item >> 3;
  const int h  = item & 7;
  const int b  = bq / LQ;

  const int HS[4] = {128, 64, 32, 16};
  const int WS[4] = {128, 64, 32, 16};
  const int ST[4] = {0, 16384, 20480, 21504};

  const float* vb = v + (size_t)b * LQ * CMODEL + h * DHEAD + lane;
  float acc = 0.f;
#pragma unroll
  for (int l = 0; l < 4; ++l) {
    const float rx = ref[((size_t)bq * 4 + l) * 2 + 0];
    const float ry = ref[((size_t)bq * 4 + l) * 2 + 1];
    const int Wl = WS[l], Hl = HS[l], s = ST[l];
#pragma unroll
    for (int p = 0; p < 4; ++p) {
      const int lp = (h * 4 + l) * 4 + p;                 // (h,l,p) flat
      const float ox = offv[(size_t)bq * 256 + lp * 2 + 0];
      const float oy = offv[(size_t)bq * 256 + lp * 2 + 1];
      const float w  = aw[(size_t)bq * 128 + lp];
      // loc = ref + off/(W,H);  x = loc.x*W - 0.5 = rx*W + ox - 0.5
      const float x = rx * (float)Wl + ox - 0.5f;
      const float y = ry * (float)Hl + oy - 0.5f;
      const float x0f = floorf(x), y0f = floorf(y);
      const float lx = x - x0f,  ly = y - y0f;
      const int x0 = (int)x0f, y0 = (int)y0f;
      const float sv =
          gatherv(vb, s, Wl, Hl, x0,     y0    ) * (1.f - lx) * (1.f - ly)
        + gatherv(vb, s, Wl, Hl, x0 + 1, y0    ) * lx         * (1.f - ly)
        + gatherv(vb, s, Wl, Hl, x0,     y0 + 1) * (1.f - lx) * ly
        + gatherv(vb, s, Wl, Hl, x0 + 1, y0 + 1) * lx         * ly;
      acc += w * sv;
    }
  }
  attn[(size_t)bq * CMODEL + h * DHEAD + lane] = f2h(acc);
}

// ---------------------------------------------------------------------------
// out = LayerNorm(a + b) * gamma + beta   (one 256-thread block per row)
// ---------------------------------------------------------------------------
__global__ __launch_bounds__(256) void add_ln_kernel(
    const float* __restrict__ a, const float* __restrict__ b,
    const float* __restrict__ gamma, const float* __restrict__ beta,
    float* __restrict__ outF, f16* __restrict__ outH)
{
  __shared__ float red[256];
  const int row = blockIdx.x;
  const int t = threadIdx.x;
  const size_t o = (size_t)row * 256 + t;
  const float x = a[o] + b[o];
  red[t] = x; __syncthreads();
#pragma unroll
  for (int s = 128; s > 0; s >>= 1) { if (t < s) red[t] += red[t + s]; __syncthreads(); }
  const float mean = red[0] * (1.f / 256.f);
  __syncthreads();
  const float d = x - mean;
  red[t] = d * d; __syncthreads();
#pragma unroll
  for (int s = 128; s > 0; s >>= 1) { if (t < s) red[t] += red[t + s]; __syncthreads(); }
  const float var = red[0] * (1.f / 256.f);
  const float y = d * rsqrtf(var + 1e-5f) * gamma[t] + beta[t];
  if (outF) outF[o] = y;
  if (outH) outH[o] = f2h(y);
}

// ---------------------------------------------------------------------------
extern "C" void kernel_launch(void* const* d_in, const int* in_sizes, int n_in,
                              void* d_out, int out_size, void* d_ws, size_t ws_size,
                              hipStream_t stream)
{
  (void)in_sizes; (void)n_in; (void)out_size; (void)ws_size;
  const float* src  = (const float*)d_in[0];
  const float* pos  = (const float*)d_in[1];
  const float* ref  = (const float*)d_in[2];
  const float* Wv   = (const float*)d_in[5];
  const float* bv   = (const float*)d_in[6];
  const float* Woff = (const float*)d_in[7];
  const float* boff = (const float*)d_in[8];
  const float* Waw  = (const float*)d_in[9];
  const float* baw  = (const float*)d_in[10];
  const float* Wout = (const float*)d_in[11];
  const float* bout = (const float*)d_in[12];
  const float* g1   = (const float*)d_in[13];
  const float* be1  = (const float*)d_in[14];
  const float* W1   = (const float*)d_in[15];
  const float* b1   = (const float*)d_in[16];
  const float* W2   = (const float*)d_in[17];
  const float* b2   = (const float*)d_in[18];
  const float* g2   = (const float*)d_in[19];
  const float* be2  = (const float*)d_in[20];

  char* ws = (char*)d_ws;
  size_t cur = 0;
  auto alloc = [&](size_t bytes) { size_t o = cur; cur += (bytes + 255) & ~(size_t)255; return o; };

  const size_t NEL = (size_t)MROWS * CMODEL;            // 11,141,120

  f16*  WvH   = (f16*)(ws + alloc(256 * 256 * sizeof(f16)));
  f16*  WoffH = (f16*)(ws + alloc(256 * 256 * sizeof(f16)));
  f16*  WawH  = (f16*)(ws + alloc(128 * 256 * sizeof(f16)));
  f16*  WoutH = (f16*)(ws + alloc(256 * 256 * sizeof(f16)));
  f16*  W1H   = (f16*)(ws + alloc(1024 * 256 * sizeof(f16)));
  f16*  W2H   = (f16*)(ws + alloc(256 * 1024 * sizeof(f16)));
  f16*  QH    = (f16*)(ws + alloc(NEL * sizeof(f16)));
  f16*  SH    = (f16*)(ws + alloc(NEL * sizeof(f16)));
  size_t oV   = alloc(NEL * sizeof(float));
  size_t oOFF = alloc(NEL * sizeof(float));
  size_t oAW  = alloc((size_t)MROWS * 128 * sizeof(float));
  f16*  ATTN  = (f16*)(ws + alloc(NEL * sizeof(f16)));
  float* X    = (float*)(ws + alloc(NEL * sizeof(float)));
  f16*  XH    = (f16*)(ws + alloc(NEL * sizeof(f16)));

  float* V    = (float*)(ws + oV);
  float* OFF  = (float*)(ws + oOFF);
  float* AW   = (float*)(ws + oAW);
  float* SRC2 = OFF;                       // reuse: offsets dead after sampling
  f16*   HID  = (f16*)(ws + oV);           // 89.1 MB over V+OFF (dead after src2)
  float* FF   = AW;                        // 44.6 MB over AW+ATTN (dead after x)
  float* OUT  = (float*)d_out;

  const dim3 gblk(32, 8);                  // 8 waves
  const int  gm = MROWS / 128;             // 340

  // 1. weights -> f16
  f32_to_f16_kernel<<<(256 * 256 + 255) / 256, 256, 0, stream>>>(Wv,   WvH,   256 * 256);
  f32_to_f16_kernel<<<(256 * 256 + 255) / 256, 256, 0, stream>>>(Woff, WoffH, 256 * 256);
  f32_to_f16_kernel<<<(128 * 256 + 255) / 256, 256, 0, stream>>>(Waw,  WawH,  128 * 256);
  f32_to_f16_kernel<<<(256 * 256 + 255) / 256, 256, 0, stream>>>(Wout, WoutH, 256 * 256);
  f32_to_f16_kernel<<<(1024 * 256 + 255) / 256, 256, 0, stream>>>(W1,  W1H,  1024 * 256);
  f32_to_f16_kernel<<<(256 * 1024 + 255) / 256, 256, 0, stream>>>(W2,  W2H,  256 * 1024);

  // 2. q = src + pos -> f16 ; src -> f16
  prep_kernel<<<(unsigned)((NEL + 255) / 256), 256, 0, stream>>>(src, pos, QH, SH, (int)NEL);

  // 3. projections (WMMA)
  gemm_f16_kernel<<<dim3(gm, 2), gblk, 0, stream>>>(SH, WvH,   bv,   V,    nullptr, MROWS, 256,  256, 0);
  gemm_f16_kernel<<<dim3(gm, 2), gblk, 0, stream>>>(QH, WoffH, boff, OFF,  nullptr, MROWS, 256,  256, 0);
  gemm_f16_kernel<<<dim3(gm, 1), gblk, 0, stream>>>(QH, WawH,  baw,  AW,   nullptr, MROWS, 128,  256, 0);

  // 4. softmax over 16 points per (b,q,h)
  softmax16_kernel<<<(MROWS * NH + 255) / 256, 256, 0, stream>>>(AW);

  // 5. bilinear sampling -> attn (f16)
  sample_kernel<<<MROWS, gblk, 0, stream>>>(V, OFF, AW, ref, ATTN);

  // 6. output projection, residual + LN1
  gemm_f16_kernel<<<dim3(gm, 2), gblk, 0, stream>>>(ATTN, WoutH, bout, SRC2, nullptr, MROWS, 256, 256, 0);
  add_ln_kernel<<<MROWS, 256, 0, stream>>>(src, SRC2, g1, be1, X, XH);

  // 7. FFN (WMMA) + residual + LN2 -> d_out
  gemm_f16_kernel<<<dim3(gm, 8), gblk, 0, stream>>>(XH,  W1H, b1, nullptr, HID, MROWS, 1024, 256,  1);
  gemm_f16_kernel<<<dim3(gm, 2), gblk, 0, stream>>>(HID, W2H, b2, FF,      nullptr, MROWS, 256, 1024, 0);
  add_ln_kernel<<<MROWS, 256, 0, stream>>>(X, FF, g2, be2, OUT, nullptr);
}
